// CALMDecoderLayer_35983236006606
// MI455X (gfx1250) — matrix operand order repaired and checked by reading, not compiled, not run on hardware
//
#include <hip/hip_runtime.h>
#include <math.h>

// ---- problem constants (match reference) ----
#define BB   4
#define TT   4
#define VV   8192
#define QQ   1024
#define CIN  32
#define COUT 16
#define KN   128          // top-K neighbourhood
#define BTN  16           // B*T

typedef _Float16 h16;
typedef __attribute__((ext_vector_type(16))) _Float16 v16h;
typedef __attribute__((ext_vector_type(8)))  float    v8f;

// Branchless tanh-form GELU: native v_exp_f32 + v_rcp_f32, overflow-safe.
__device__ __forceinline__ float gelu_f(float x) {
  float t  = 0.7978845608028654f * x * (1.0f + 0.044715f * x * x);
  float a  = __expf(-2.0f * fabsf(t));                       // (0,1]
  float th = (1.0f - a) * __builtin_amdgcn_rcpf(1.0f + a);   // |tanh(t)|
  th = copysignf(th, t);
  return 0.5f * x * (1.0f + th);
}
__device__ __forceinline__ float wrap_unit(float d) {
  d += 0.5f; d -= floorf(d); return d - 0.5f;
}

// ---- WMMA operand loaders (CDNA5 wave32 layouts, CDNA5 ISA §7.12.2) ----
// A: 16x32 f16, row-major LDS tile, leading dim ld (halves). Two b128 loads/lane.
__device__ __forceinline__ v16h lds_load_A16x32(const h16* base, int ld, int lane) {
  int row = lane & 15;
  int klo = (lane < 16) ? 0 : 16;
  const h16* p = base + row * ld;
  v16h a;
#pragma unroll
  for (int i = 0; i < 8; ++i) { a[i] = p[klo + i]; a[8 + i] = p[klo + 8 + i]; }
  return a;
}
// B: 32x16 f16 tile stored pre-swizzled lane-major (32 lanes x 16 contiguous
// halves). Element (row j, col d) lives at lane (d + (j&16?16:0)), slot (j&15).
// Per-lane read = 2 contiguous b128 loads.
__device__ __forceinline__ v16h lds_load_Bsw(const h16* base, int lane) {
  const int4* p = (const int4*)(base + lane * 16);
  v16h b;
  int4* bp = (int4*)&b;
  bp[0] = p[0]; bp[1] = p[1];
  return b;
}
// C/D v8f: element r -> M = r + (lane<16?0:8), N = lane%16.

// swizzled offset for B element (j = contraction row, d = col) within a tile
__device__ __host__ __forceinline__ int bsw_off(int j, int d) {
  return ((j & 16) << 4) + d * 16 + (j & 15);
}

// ---------------------------------------------------------------------------
// Kernel 0: convert W1/W2 to f16 in B-swizzled tile layout; emit query_pos tail.
// W1sw: 2 tiles of 512 halves; W2sw: 32 tiles of 512 halves.
// ---------------------------------------------------------------------------
__global__ __launch_bounds__(256) void prep_weights(
    const float* __restrict__ W1, const float* __restrict__ W2,
    const float* __restrict__ qpos, h16* __restrict__ W1sw, h16* __restrict__ W2sw,
    float* __restrict__ out_tail) {
  int tid = threadIdx.x;
  for (int i = tid; i < 32 * 32; i += 256) {
    int j = i >> 5, n = i & 31;
    int nt = n >> 4, d = n & 15;
    W1sw[nt * 512 + bsw_off(j, d)] = (h16)W1[i];
  }
  for (int i = tid; i < 32 * 512; i += 256) {
    int j = i >> 9, n = i & 511;
    int nt = n >> 4, d = n & 15;
    W2sw[nt * 512 + bsw_off(j, d)] = (h16)W2[i];
  }
  for (int i = tid; i < QQ * 2; i += 256) out_tail[i] = qpos[i];
}

// ---------------------------------------------------------------------------
// Kernel 1: xl = x @ W_lin + b_lin  -> f16  (memory bound: 16MB in, 8MB out)
// ---------------------------------------------------------------------------
__global__ __launch_bounds__(256) void xl_kernel(
    const float* __restrict__ x, const float* __restrict__ Wl,
    const float* __restrict__ bl, h16* __restrict__ xlh) {
  __shared__ float sW[32 * 32];
  __shared__ float sB[32];
  __shared__ float sX[8 * 32];
  int tid = threadIdx.x;
  for (int i = tid; i < 1024; i += 256) sW[i] = Wl[i];
  if (tid < 32) sB[tid] = bl[tid];
  long rowbase = (long)blockIdx.x * 8;
  sX[tid] = x[rowbase * 32 + tid];
  __syncthreads();
  int r = tid >> 5, c = tid & 31;
  float acc = sB[c];
#pragma unroll
  for (int j = 0; j < 32; ++j) acc += sX[r * 32 + j] * sW[j * 32 + c];
  xlh[(rowbase + r) * 32 + c] = (h16)acc;
}

// ---------------------------------------------------------------------------
// Kernel 2: per-query top-128 via 4-pass radix select on float bits (LDS),
// then normalized-distance softmax weights. One 256-thread block per query.
// ---------------------------------------------------------------------------
__global__ __launch_bounds__(256) void topk_kernel(
    const float* __restrict__ pos, const float* __restrict__ qpos,
    int* __restrict__ ind, float* __restrict__ dist, float* __restrict__ kdist) {
  __shared__ float     se[VV];       // 32 KB distance array
  __shared__ unsigned  hist[256];
  __shared__ float     sel_e[KN];
  __shared__ int       sel_i[KN];
  __shared__ float     red[KN];
  __shared__ unsigned  sprefix;
  __shared__ int       srem;
  __shared__ unsigned  scnt;
  int q = blockIdx.x, tid = threadIdx.x;
  float qx = qpos[2 * q], qy = qpos[2 * q + 1];
  for (int v = tid; v < VV; v += 256) {
    float dx = wrap_unit(qx - pos[2 * v]);
    float dy = wrap_unit(qy - pos[2 * v + 1]);
    se[v] = dx * dx + dy * dy;
  }
  if (tid == 0) { sprefix = 0u; srem = KN - 1; scnt = 0u; }
  __syncthreads();
  // exact 128th-smallest key via radix select (positive floats: bit order == value order)
  for (int p = 0; p < 4; ++p) {
    int shift = 24 - 8 * p;
    unsigned himask = p ? (0xFFFFFFFFu << (32 - 8 * p)) : 0u;
    unsigned pref = sprefix;
    hist[tid] = 0u;
    __syncthreads();
    for (int v = tid; v < VV; v += 256) {
      unsigned key = __float_as_uint(se[v]);
      if ((key & himask) == pref) atomicAdd(&hist[(key >> shift) & 255u], 1u);
    }
    __syncthreads();
    if (tid == 0) {
      int rem = srem; unsigned run = 0;
      for (int d = 0; d < 256; ++d) {
        unsigned c = hist[d];
        if (run + c > (unsigned)rem) {
          sprefix = pref | ((unsigned)d << shift);
          srem = rem - (int)run;
          break;
        }
        run += c;
      }
    }
    __syncthreads();
  }
  unsigned kth = sprefix;
  for (int v = tid; v < VV; v += 256) {
    unsigned key = __float_as_uint(se[v]);
    if (key < kth) { unsigned s = atomicAdd(&scnt, 1u); sel_i[s] = v; sel_e[s] = se[v]; }
  }
  __syncthreads();
  for (int v = tid; v < VV; v += 256) {
    unsigned key = __float_as_uint(se[v]);
    if (key == kth) {
      unsigned s = atomicAdd(&scnt, 1u);
      if (s < KN) { sel_i[s] = v; sel_e[s] = se[v]; }
    }
  }
  __syncthreads();
  // edist min/max normalize + softmax(-edist) over K (TEMP=1)
  float ek = (tid < KN) ? sel_e[tid] : 0.f;
  if (tid < KN) red[tid] = ek;
  __syncthreads();
  for (int s = KN / 2; s > 0; s >>= 1) { if (tid < s) red[tid] = fminf(red[tid], red[tid + s]); __syncthreads(); }
  float mn = red[0];
  __syncthreads();
  float ed = ek - mn;
  if (tid < KN) red[tid] = ed;
  __syncthreads();
  for (int s = KN / 2; s > 0; s >>= 1) { if (tid < s) red[tid] = fmaxf(red[tid], red[tid + s]); __syncthreads(); }
  float mx = red[0];
  __syncthreads();
  float w = __expf(-(ed / (mx + 1e-8f)));
  if (tid < KN) red[tid] = w;
  __syncthreads();
  for (int s = KN / 2; s > 0; s >>= 1) { if (tid < s) red[tid] += red[tid + s]; __syncthreads(); }
  float sum = red[0];
  if (tid < KN) {
    int v = sel_i[tid];
    ind[q * KN + tid]   = v;
    kdist[q * KN + tid] = w / sum;
    float dx = wrap_unit(qx - pos[2 * v]);
    float dy = wrap_unit(qy - pos[2 * v + 1]);
    dist[q * 2 * KN + 2 * tid]     = dx;
    dist[q * 2 * KN + 2 * tid + 1] = dy;
  }
}

// ---------------------------------------------------------------------------
// Kernel 3: fused per-query WMMA pipeline. One 256-thread block (8 waves) / query.
// ~131 KB dynamic LDS => 2 workgroups / 320KB WGP.
// ---------------------------------------------------------------------------
#define OFF_W2   0        // 32 swizzled tiles (32768 B)
#define OFF_W1   32768    // 2 swizzled tiles (2048 B)
#define OFF_H    34816    // 128x32 h16 (8192 B)
#define OFF_RFF  43008    // 128x32 h16 (8192 B)
#define OFF_KER  51200    // 32 swizzled tiles (32768 B)
#define OFF_XG   83968    // 16x1024 h16 (32768 B)
#define OFF_YP   116736   // 8x256 f32 (8192 B)
#define OFF_Y    124928   // 256 f32 (1024 B)
#define OFF_HID  125952   // 16x64 f32 (4096 B)
#define OFF_DIST 130048   // 128x2 f32 (1024 B)
#define OFF_KD   131072   // 128 f32 (512 B)
#define OFF_IND  131584   // 128 i32 (512 B)
#define OFF_QM   132096   // 32 f32
#define OFF_QO   132224   // 32 f32
#define OFF_FL   132352   // 512 f32 (2048 B)
#define OFF_B1   134400   // 32 f32
#define OFF_BR   134528   // 32 f32
#define SMEM_FUSED 134656

__global__ __launch_bounds__(256) void fused_kernel(
    const h16* __restrict__ xlh, const int* __restrict__ indg,
    const float* __restrict__ distg, const float* __restrict__ kdistg,
    const h16* __restrict__ W1sw, const h16* __restrict__ W2sw,
    const float* __restrict__ b1, const float* __restrict__ qmod_w,
    const float* __restrict__ qmod_off, const float* __restrict__ Brff,
    const float* __restrict__ filt, const float* __restrict__ bias_out,
    const float* __restrict__ Wm1, const float* __restrict__ bm1,
    const float* __restrict__ Wm2, const float* __restrict__ bm2,
    float* __restrict__ out) {
  extern __shared__ __align__(16) char smem[];
  h16*   sW2  = (h16*)(smem + OFF_W2);
  h16*   sW1  = (h16*)(smem + OFF_W1);
  h16*   sH   = (h16*)(smem + OFF_H);
  h16*   sRf  = (h16*)(smem + OFF_RFF);
  h16*   sKer = (h16*)(smem + OFF_KER);
  h16*   sXg  = (h16*)(smem + OFF_XG);
  float* sYp  = (float*)(smem + OFF_YP);
  float* sY   = (float*)(smem + OFF_Y);
  float* sHid = (float*)(smem + OFF_HID);
  float* sDist= (float*)(smem + OFF_DIST);
  float* sKd  = (float*)(smem + OFF_KD);
  int*   sInd = (int*)  (smem + OFF_IND);
  float* sQm  = (float*)(smem + OFF_QM);
  float* sQo  = (float*)(smem + OFF_QO);
  float* sFl  = (float*)(smem + OFF_FL);
  float* sB1  = (float*)(smem + OFF_B1);
  float* sBr  = (float*)(smem + OFF_BR);

  int q = blockIdx.x, tid = threadIdx.x;
  int wave = tid >> 5, lane = tid & 31;
  int hi8 = (lane < 16) ? 0 : 8;
  int cl  = lane & 15;

  // ---- cooperative loads ----
  { const int4* src = (const int4*)W2sw; int4* dst = (int4*)sW2;
    for (int i = tid; i < 2048; i += 256) dst[i] = src[i]; }
  { const int4* src = (const int4*)W1sw; int4* dst = (int4*)sW1;
    if (tid < 128) dst[tid] = src[tid]; }
  if (tid < 32) { sQm[tid] = qmod_w[q * 32 + tid]; sQo[tid] = qmod_off[q * 32 + tid];
                  sB1[tid] = b1[tid];              sBr[tid] = Brff[tid]; }
  for (int i = tid; i < 512; i += 256) sFl[i] = filt[i];
  if (tid < KN) { sKd[tid] = kdistg[q * KN + tid]; sInd[tid] = indg[q * KN + tid]; }
  sDist[tid] = distg[q * 2 * KN + tid];   // 256 floats
  __syncthreads();

  // ---- random Fourier features (128x32 f16), native sin/cos ----
  for (int job = tid; job < KN * 16; job += 256) {
    int k = job >> 4, j = job & 15;
    float px = 6.28318530717958647692f *
               (sDist[2 * k] * sBr[j] + sDist[2 * k + 1] * sBr[16 + j]);
    sRf[k * 32 + j]      = (h16)__sinf(px);
    sRf[k * 32 + 16 + j] = (h16)__cosf(px);
  }
  __syncthreads();

  // ---- GEMM1: h = gelu((rff @ W1 + b1) * qmod + qoff), 128x32 ----
  for (int t = wave; t < 16; t += 8) {
    int mt = t >> 1, nt = t & 1;
    v16h a = lds_load_A16x32(sRf + mt * 16 * 32, 32, lane);
    v16h b = lds_load_Bsw(sW1 + nt * 512, lane);
    int col = nt * 16 + cl;
    float bv = sB1[col];
    v8f c;
#pragma unroll
    for (int r = 0; r < 8; ++r) c[r] = bv;
    c = __builtin_amdgcn_wmma_f32_16x16x32_f16(false, a, false, b, (short)0, c, false, false);
    float qm = sQm[col], qo = sQo[col];
#pragma unroll
    for (int r = 0; r < 8; ++r) {
      int row = mt * 16 + r + hi8;
      sH[row * 32 + col] = (h16)gelu_f(c[r] * qm + qo);
    }
  }
  __syncthreads();

  // ---- main loop: 4 chunks of 32 neighbours ----
  v8f acc = {};
  for (int kc = 0; kc < 4; ++kc) {
    // GEMM2: ker chunk = (h@W2 + filt) * kdist, written B-swizzled into sKer.
    // Each wave: 2 A tiles, reused across its 4 N tiles.
#pragma unroll
    for (int mt = 0; mt < 2; ++mt) {
      v16h a = lds_load_A16x32(sH + (kc * 32 + mt * 16) * 32, 32, lane);
#pragma unroll
      for (int i = 0; i < 4; ++i) {
        int nt = wave + i * 8;
        v16h b = lds_load_Bsw(sW2 + nt * 512, lane);
        v8f c = {};
        c = __builtin_amdgcn_wmma_f32_16x16x32_f16(false, a, false, b, (short)0, c, false, false);
        int col = nt * 16 + cl;            // 0..511 = cc*16 + dd
        int ccu = col >> 4, ddu = col & 15;
        int swoff = bsw_off(ccu, ddu);     // uniform over r
        float fl = sFl[col];
#pragma unroll
        for (int r = 0; r < 8; ++r) {
          int kk = mt * 16 + r + hi8;
          sKer[kk * 512 + swoff] = (h16)((c[r] + fl) * sKd[kc * 32 + kk]);
        }
      }
    }
    // gather Xg chunk: sXg[bt][kk*32+c] = xl[bt, ind[q, kc*32+kk], c]
    for (int job = tid; job < 512; job += 256) {
      int bt = job >> 5, kk = job & 31;
      int v = sInd[kc * 32 + kk];
      const int4* src = (const int4*)(xlh + ((size_t)bt * VV + v) * 32);
      int4* dst = (int4*)(sXg + bt * 1024 + kk * 32);
#pragma unroll
      for (int i = 0; i < 4; ++i) dst[i] = src[i];
    }
    __syncthreads();
    // GEMM3: y(16x16) += Xg(16x1024) @ Ker'(1024x16); K split over 8 waves
    for (int kt = wave; kt < 32; kt += 8) {
      v16h a = lds_load_A16x32(sXg + kt * 32, 1024, lane);
      v16h b = lds_load_Bsw(sKer + kt * 512, lane);   // swizzled kk-tile
      acc = __builtin_amdgcn_wmma_f32_16x16x32_f16(false, a, false, b, (short)0, acc, false, false);
    }
    __syncthreads();
  }

  // ---- reduce per-wave partials ----
#pragma unroll
  for (int r = 0; r < 8; ++r) sYp[wave * 256 + (r + hi8) * 16 + cl] = acc[r];
  __syncthreads();
  int bt = tid >> 4, dd = tid & 15;
  float ysum = 0.f;
#pragma unroll
  for (int w = 0; w < 8; ++w) ysum += sYp[w * 256 + tid];
  float yg = gelu_f(ysum + bias_out[dd]);
  sY[tid] = yg;
  __syncthreads();

  // ---- residual MLP: gelu(yg@Wm1+bm1)@Wm2+bm2 + yg ----
  for (int job = tid; job < BTN * 64; job += 256) {
    int bb = job >> 6, j = job & 63;
    float a2 = bm1[j];
#pragma unroll
    for (int d2 = 0; d2 < 16; ++d2) a2 += sY[bb * 16 + d2] * Wm1[d2 * 64 + j];
    sHid[bb * 64 + j] = gelu_f(a2);
  }
  __syncthreads();
  float o = bm2[dd];
#pragma unroll
  for (int j = 0; j < 64; ++j) o += sHid[bt * 64 + j] * Wm2[j * 16 + dd];
  o += sY[tid];
  out[((size_t)bt * QQ + q) * COUT + dd] = o;
}

// ---------------------------------------------------------------------------
// Host launcher
// ---------------------------------------------------------------------------
extern "C" void kernel_launch(void* const* d_in, const int* in_sizes, int n_in,
                              void* d_out, int out_size, void* d_ws, size_t ws_size,
                              hipStream_t stream) {
  const float* x        = (const float*)d_in[0];
  const float* pos      = (const float*)d_in[1];
  const float* qpos     = (const float*)d_in[2];
  const float* qmod_w   = (const float*)d_in[3];
  const float* qmod_off = (const float*)d_in[4];
  const float* W_lin    = (const float*)d_in[5];
  const float* b_lin    = (const float*)d_in[6];
  const float* B_rff    = (const float*)d_in[7];
  const float* W1       = (const float*)d_in[8];
  const float* b1       = (const float*)d_in[9];
  const float* W2       = (const float*)d_in[10];
  const float* filt     = (const float*)d_in[11];
  const float* bias_out = (const float*)d_in[12];
  const float* Wm1      = (const float*)d_in[13];
  const float* bm1      = (const float*)d_in[14];
  const float* Wm2      = (const float*)d_in[15];
  const float* bm2      = (const float*)d_in[16];
  float* out = (float*)d_out;

  // workspace layout
  char* ws = (char*)d_ws;
  h16*   xlh   = (h16*)  (ws + 0);                    //  8,388,608 B
  h16*   W1sw  = (h16*)  (ws + 8388608);              //      2,048 B
  h16*   W2sw  = (h16*)  (ws + 8390656);              //     32,768 B
  int*   ind   = (int*)  (ws + 8423424);              //    524,288 B
  float* dist  = (float*)(ws + 8947712);              //  1,048,576 B
  float* kdist = (float*)(ws + 9996288);              //    524,288 B

  float* out_tail = out + (size_t)BB * TT * QQ * COUT;   // query_pos passthrough

  prep_weights<<<1, 256, 0, stream>>>(W1, W2, qpos, W1sw, W2sw, out_tail);
  xl_kernel<<<(BB * TT * VV) / 8, 256, 0, stream>>>(x, W_lin, b_lin, xlh);
  topk_kernel<<<QQ, 256, 0, stream>>>(pos, qpos, ind, dist, kdist);
  fused_kernel<<<QQ, 256, SMEM_FUSED, stream>>>(
      xlh, ind, dist, kdist, W1sw, W2sw, b1, qmod_w, qmod_off, B_rff,
      filt, bias_out, Wm1, bm1, Wm2, bm2, out);
}
